// EquivariantProductBasisBlock_27273042330415
// MI455X (gfx1250) — matrix operand order repaired and checked by reading, not compiled, not run on hardware
//
#include <hip/hip_runtime.h>

// ---------------------------------------------------------------------------
// MACE EquivariantProductBasisBlock on gfx1250 (MI455X), bf16 WMMA formulation
// with ASYNCcnt-pipelined global->LDS staging of the constant B matrix.
// ---------------------------------------------------------------------------

typedef __attribute__((ext_vector_type(16))) __bf16 v16bf;
typedef __attribute__((ext_vector_type(16))) float  v16f;
typedef __attribute__((ext_vector_type(8)))  float  v8f;

static constexpr int N_NODES = 1024;
static constexpr int C       = 128;
static constexpr int LM      = 16;
static constexpr int ELEMS   = 10;
static constexpr int P1 = 4, P2 = 12, P3 = 30;
static constexpr int KSTEPS  = 17;            // 16 m-steps (p3 padded to 32) + 1 nu2-step
static constexpr int NCT     = 64;            // 1024 Lij columns / 16
static constexpr int NC      = N_NODES * C;   // 131072 (n,c) rows

static constexpr int      PHASE_TILES = 4;    // column tiles staged per phase
static constexpr int      NPHASE      = NCT / PHASE_TILES;          // 16
static constexpr unsigned TILE_BYTES  = KSTEPS * 32 * 32;           // 17,408
static constexpr unsigned PHASE_BYTES = PHASE_TILES * TILE_BYTES;   // 69,632
static constexpr unsigned SMEM_BYTES  = 2 * PHASE_BYTES + 8 * 512 * 4; // 155,648

// workspace layout (bytes)
static constexpr size_t BFRAG_BYTES = (size_t)KSTEPS * NCT * 32 * 16 * 2; // 1,114,112
static constexpr size_t B1F_OFF     = BFRAG_BYTES;                        // f32 [10][64][128]
static constexpr size_t B1F_BYTES   = (size_t)ELEMS * 64 * C * 4;         // 327,680
static constexpr size_t F_OFF       = B1F_OFF + B1F_BYTES;                // f32 [NC][4]

// ---------------------------------------------------------------------------
// Prep 1: WMMA-fragment-ready B matrix (bf16) from U3 / U2, tile-major so each
// 4-tile phase is one contiguous 69,632B block for the async LDS copy.
// Layout: bfrag[((ct*17 + t)*32 + lane)*16 + q]
//   B-operand k order used here, same as A (16x16x32 bf16): lane -> column N = lane&15,
//   element q -> K = ((lane>>4)<<4) + q; logical k = 32*t + K, m = t, p = K.
//   Step t==16 carries the nu=2 basis: value = U2[L,i,j,p] for p<12.
// ---------------------------------------------------------------------------
__global__ void prep_bfrag_kernel(const float* __restrict__ U2_0, const float* __restrict__ U3_0,
                                  const float* __restrict__ U2_1, const float* __restrict__ U3_1,
                                  __bf16* __restrict__ bfrag) {
  int tid = blockIdx.x * blockDim.x + threadIdx.x;   // 64*17*32*16 = 557056
  if (tid >= NCT * KSTEPS * 32 * 16) return;
  int q    = tid & 15;
  int lane = (tid >> 4) & 31;
  int rest = tid >> 9;                               // 0..1087
  int t    = rest % KSTEPS;
  int ct   = rest / KSTEPS;
  int K    = ((lane >> 4) << 4) + q;                 // 0..31
  int col  = ct * 16 + (lane & 15);                  // 0..1023
  int L, i, j;
  const float* U3;
  const float* U2;
  if (col < 256) { L = 0; i = (col >> 4) & 15; j = col & 15; U3 = U3_0; U2 = U2_0; }
  else { int cc = col - 256; L = cc >> 8; i = (cc >> 4) & 15; j = cc & 15; U3 = U3_1; U2 = U2_1; }
  float v = 0.0f;
  if (t < 16) { if (K < P3) v = U3[(((L * 16 + i) * 16 + j) * 16 + t) * P3 + K]; }
  else        { if (K < P2) v = U2[((L * 16 + i) * 16 + j) * P2 + K]; }
  bfrag[tid] = static_cast<__bf16>(v);
}

// ---------------------------------------------------------------------------
// Prep 2: fused nu=1 bias B1f[e][Li][c] = sum_p U1[L,i,p] * W1[e,p,c]
// ---------------------------------------------------------------------------
__global__ void prep_b1f_kernel(const float* __restrict__ U1_0, const float* __restrict__ W1_0,
                                const float* __restrict__ U1_1, const float* __restrict__ W1_1,
                                float* __restrict__ B1f) {
  int tid = blockIdx.x * blockDim.x + threadIdx.x;   // 10*64*128 = 81920
  if (tid >= ELEMS * 64 * C) return;
  int c  = tid & 127;
  int Li = (tid >> 7) & 63;
  int e  = tid >> 13;
  float s = 0.0f;
  if (Li < 16) {
    int i = Li;
    #pragma unroll
    for (int p = 0; p < P1; ++p) s += U1_0[i * P1 + p] * W1_0[(e * P1 + p) * C + c];
  } else {
    int qq = Li - 16; int L = qq >> 4, i = qq & 15;
    #pragma unroll
    for (int p = 0; p < P1; ++p) s += U1_1[(L * 16 + i) * P1 + p] * W1_1[(e * P1 + p) * C + c];
  }
  B1f[tid] = s;
}

// ---------------------------------------------------------------------------
// Main symmetric-contraction kernel. One wave owns 16 (n,c) rows; the 8 waves
// of a block share B fragments staged in LDS by async global->LDS loads.
// Rolled phase loops: per tile ct = 4f+ctl,
//   Li == ct, it == 4*(f&3)+ctl, fidx == f>>2 (uniform per phase).
// Inner K-loop: 3-deep DS-load pipeline enforced via sched_group_barrier.
// ---------------------------------------------------------------------------
__global__ __launch_bounds__(256)
void sym_contract_kernel(const float* __restrict__ x, const int* __restrict__ species,
                         const float* __restrict__ W2_0, const float* __restrict__ W3_0,
                         const float* __restrict__ W2_1, const float* __restrict__ W3_1,
                         const __bf16* __restrict__ bfrag, const float* __restrict__ B1f,
                         float* __restrict__ F) {
  extern __shared__ char dynsmem[];                  // 2*PHASE_BYTES + 16KB transpose
  const int wv   = threadIdx.x >> 5;
  const int l    = threadIdx.x & 31;
  const int rb   = blockIdx.x * 8 + wv;              // row-block (16 rows)
  const int M    = l & 15;
  const int half = l >> 4;
  const int r    = rb * 16 + M;
  const int n    = r >> 7;
  const int c    = r & 127;
  const int e    = species[n];
  const unsigned lds_lo = (unsigned)(size_t)dynsmem; // LDS byte address of base

  float xr[16];
  const float* xp = x + (size_t)r * LM;
  #pragma unroll
  for (int t = 0; t < 16; ++t) xr[t] = xp[t];

  float* tile0 = (float*)(dynsmem + 2 * PHASE_BYTES) + wv * 512;
  v16bf afrag[KSTEPS];
  float fa0 = 0.f, fa1 = 0.f, fa2 = 0.f, fa3 = 0.f;

  // Issue one phase (4 tiles, 69,632B) of async global->LDS copies.
  auto issue_phase = [&](int ph) {
    #pragma unroll
    for (int jj = 0; jj < 17; ++jj) {
      unsigned chunk = (threadIdx.x + jj * 256) * 16u;
      unsigned goff  = (unsigned)ph * PHASE_BYTES + chunk;
      unsigned loff  = lds_lo + (unsigned)(ph & 1) * PHASE_BYTES + chunk;
      asm volatile("global_load_async_to_lds_b128 %0, %1, %2"
                   :: "v"(loff), "v"(goff), "s"(bfrag) : "memory");
    }
  };

  // Build the 17 A fragments for one irrep block: A[row,32t+p] = x[row,t]*W3[e,p,c]
  auto build_a = [&](const float* W3, const float* W2) {
    v16f w3vec, w2vec;
    #pragma unroll
    for (int q = 0; q < 16; ++q) {
      int p = (half << 4) + q;
      w3vec[q] = (p < P3) ? W3[(e * P3 + p) * C + c] : 0.0f;
      w2vec[q] = (p < P2) ? W2[(e * P2 + p) * C + c] : 0.0f;
    }
    #pragma unroll
    for (int t = 0; t < 16; ++t) {
      v16f prod = w3vec * xr[t];
      afrag[t] = __builtin_convertvector(prod, v16bf);   // v_cvt_pk_bf16_f32
    }
    afrag[16] = __builtin_convertvector(w2vec, v16bf);
  };

  // Compute the 4 column tiles of phase f from LDS; returns sum of S'*x_i
  // contributions (all tiles of a phase feed the same output component).
  auto compute_phase = [&](int f) -> float {
    const v16bf* lb = (const v16bf*)(dynsmem + (f & 1) * PHASE_BYTES);
    const int sel = f & 3;
    float facc = 0.f;
    #pragma unroll
    for (int ctl = 0; ctl < PHASE_TILES; ++ctl) {
      const v16bf* lbt = lb + (ctl * KSTEPS) * 32 + l;   // fragment t at lbt[32*t]
      // 3-deep software pipeline: DS latency hidden behind ~3 WMMAs.
      v16bf b0 = lbt[0];
      v16bf b1 = lbt[32];
      v16bf b2 = lbt[64];
      v8f acc = {0.f, 0.f, 0.f, 0.f, 0.f, 0.f, 0.f, 0.f};
      #pragma unroll
      for (int t = 0; t < KSTEPS; ++t) {
        v16bf bn = b2;
        if (t + 3 < KSTEPS) bn = lbt[(t + 3) * 32];
        acc = __builtin_amdgcn_wmma_f32_16x16x32_bf16(false, afrag[t], false, b0,
                                                      (short)0, acc, false, false);
        b0 = b1; b1 = b2; b2 = bn;
      }
      // IGLP pattern for this tile's K-section: preload 3 fragments (6 b128
      // DS reads), then 14x [1 WMMA, 2 DS reads], then 3 draining WMMAs.
      __builtin_amdgcn_sched_group_barrier(0x100, 6, 0);      // DS read x6
      #pragma unroll
      for (int t = 0; t < KSTEPS - 3; ++t) {
        __builtin_amdgcn_sched_group_barrier(0x008, 1, 0);    // WMMA
        __builtin_amdgcn_sched_group_barrier(0x100, 2, 0);    // DS read x2
      }
      __builtin_amdgcn_sched_group_barrier(0x008, 3, 0);      // WMMA drain

      // Transpose accumulator tile through wave-private LDS (double-buffered).
      float* tile = tile0 + (ctl & 1) * 256;
      #pragma unroll
      for (int kk = 0; kk < 8; ++kk) {
        int Mk = kk + half * 8;                      // row of this element
        tile[Mk * 16 + M] = acc[kk];                 // column j = lane&15
      }
      const float4* rp4 = (const float4*)(tile + M * 16);
      float S = 0.0f;
      #pragma unroll
      for (int jj = 0; jj < 4; ++jj) {
        float4 rv = rp4[jj];
        S += rv.x * xr[jj * 4 + 0] + rv.y * xr[jj * 4 + 1] +
             rv.z * xr[jj * 4 + 2] + rv.w * xr[jj * 4 + 3];
      }
      int ct = f * PHASE_TILES + ctl;                // Li == ct
      S += B1f[(e * 64 + ct) * C + c];
      float xit = (sel == 0) ? xr[ctl] : (sel == 1) ? xr[4 + ctl]
                : (sel == 2) ? xr[8 + ctl] : xr[12 + ctl];   // x[row, 4*(f&3)+ctl]
      facc += S * xit;
      // Account for the epilogue DS ops so the K-pattern cannot absorb them.
      __builtin_amdgcn_sched_group_barrier(0x200, 8, 0);      // DS write x8
      __builtin_amdgcn_sched_group_barrier(0x100, 4, 0);      // DS read x4
    }
    return facc;
  };

  issue_phase(0);

  // ---- irrep block 0 (d=1): phases 0..3, output component fa0 ----
  build_a(W3_0, W2_0);
  #pragma unroll 1
  for (int f = 0; f < 4; ++f) {
    asm volatile("s_wait_asynccnt 0" ::: "memory");  // own async loads done
    __syncthreads();                                 // whole block's loads done
    issue_phase(f + 1);                              // prefetch next phase
    fa0 += compute_phase(f);
  }

  // ---- irrep block 1 (d=3): phases 4..15, components fa1..fa3 ----
  build_a(W3_1, W2_1);
  #pragma unroll 1
  for (int f = 4; f < NPHASE; ++f) {
    asm volatile("s_wait_asynccnt 0" ::: "memory");
    __syncthreads();
    if (f + 1 < NPHASE) issue_phase(f + 1);
    float facc = compute_phase(f);
    int fidx = f >> 2;                               // 1..3, uniform per phase
    fa1 += (fidx == 1) ? facc : 0.f;
    fa2 += (fidx == 2) ? facc : 0.f;
    fa3 += (fidx == 3) ? facc : 0.f;
  }

  if (l < 16) {
    float* fp = F + (size_t)r * 4;
    fp[0] = fa0; fp[1] = fa1; fp[2] = fa2; fp[3] = fa3;
  }
}

// ---------------------------------------------------------------------------
// Final linear: y0 = F0 @ Wlin0 * s ; y1[:, :, i] = F1[:, :, i] @ Wlin1 * s.
// 4 GEMMs of 1024x128x128, bf16 WMMA, K = 4 steps of 32.
// ---------------------------------------------------------------------------
__global__ __launch_bounds__(256)
void linear_kernel(const float* __restrict__ F, const float* __restrict__ Wlin0,
                   const float* __restrict__ Wlin1, float* __restrict__ out) {
  const int wv    = threadIdx.x >> 5;
  const int l     = threadIdx.x & 31;
  const int w     = blockIdx.x * 8 + wv;             // 0..2047
  const int which = w & 3;                           // 0: y0, 1..3: y1 i=which-1
  const int ctile = (w >> 2) & 7;
  const int ntile = w >> 5;                          // 0..63
  const int half  = l >> 4;
  const int Ml    = l & 15;
  const int kcol  = ctile * 16 + Ml;
  const float* Wl = which ? Wlin1 : Wlin0;

  v8f acc = {0.f, 0.f, 0.f, 0.f, 0.f, 0.f, 0.f, 0.f};
  #pragma unroll
  for (int s = 0; s < 4; ++s) {
    v16f af, bf;
    #pragma unroll
    for (int q = 0; q < 16; ++q) {
      int Ka = (half << 4) + q;
      int ca = s * 32 + Ka;                          // channel index (A reduction)
      int nrow = ntile * 16 + Ml;
      af[q] = F[((size_t)(nrow * C + ca)) * 4 + which];
      int Kb = (half << 4) + q;
      int cb = s * 32 + Kb;
      bf[q] = Wl[cb * C + kcol];
    }
    v16bf a = __builtin_convertvector(af, v16bf);
    v16bf b = __builtin_convertvector(bf, v16bf);
    acc = __builtin_amdgcn_wmma_f32_16x16x32_bf16(false, a, false, b,
                                                  (short)0, acc, false, false);
  }
  const float scale = 0.08838834764831845f;          // 1/sqrt(128)
  #pragma unroll
  for (int kk = 0; kk < 8; ++kk) {
    int Mk = kk + half * 8;
    int nrow = ntile * 16 + Mk;
    int oidx = (which == 0) ? nrow * 512 + kcol
                            : nrow * 512 + 128 + kcol * 3 + (which - 1);
    out[oidx] = acc[kk] * scale;
  }
}

// ---------------------------------------------------------------------------
extern "C" void kernel_launch(void* const* d_in, const int* in_sizes, int n_in,
                              void* d_out, int out_size, void* d_ws, size_t ws_size,
                              hipStream_t stream) {
  (void)in_sizes; (void)n_in; (void)out_size; (void)ws_size;
  const float* x       = (const float*)d_in[0];
  const int*   species = (const int*)d_in[1];
  const float* U1_0 = (const float*)d_in[2];
  const float* U2_0 = (const float*)d_in[3];
  const float* U3_0 = (const float*)d_in[4];
  const float* W1_0 = (const float*)d_in[5];
  const float* W2_0 = (const float*)d_in[6];
  const float* W3_0 = (const float*)d_in[7];
  const float* Wl_0 = (const float*)d_in[8];
  const float* U1_1 = (const float*)d_in[9];
  const float* U2_1 = (const float*)d_in[10];
  const float* U3_1 = (const float*)d_in[11];
  const float* W1_1 = (const float*)d_in[12];
  const float* W2_1 = (const float*)d_in[13];
  const float* W3_1 = (const float*)d_in[14];
  const float* Wl_1 = (const float*)d_in[15];
  float* out = (float*)d_out;

  __bf16* bfrag = (__bf16*)d_ws;
  float*  B1f   = (float*)((char*)d_ws + B1F_OFF);
  float*  F     = (float*)((char*)d_ws + F_OFF);

  prep_bfrag_kernel<<<(NCT * KSTEPS * 32 * 16 + 255) / 256, 256, 0, stream>>>(
      U2_0, U3_0, U2_1, U3_1, bfrag);
  prep_b1f_kernel<<<(ELEMS * 64 * C + 255) / 256, 256, 0, stream>>>(
      U1_0, W1_0, U1_1, W1_1, B1f);
  sym_contract_kernel<<<NC / 128, 256, SMEM_BYTES, stream>>>(
      x, species, W2_0, W3_0, W2_1, W3_1, bfrag, B1f, F);
  linear_kernel<<<256, 256, 0, stream>>>(F, Wl_0, Wl_1, out);
}
